// EncoderLayer_34333968564745
// MI455X (gfx1250) — matrix operand order repaired and checked by reading, not compiled, not run on hardware
//
#include <hip/hip_runtime.h>
#include <hip/hip_bf16.h>

typedef __attribute__((ext_vector_type(16))) _Float16 v16h;
typedef __attribute__((ext_vector_type(8)))  float    v8f;

#define N_NODES 50000
#define N_EDGES 800000
#define IN_F    128
#define OUT_F   256
#define M_TILES 3125          // 50000 / 16
#define EPS     1e-5f

// ---------------- zero init (h + stat accumulators) ----------------
__global__ void zero_kernel(float* p, int n) {
    int i = blockIdx.x * 256 + threadIdx.x;
    if (i < n) p[i] = 0.0f;
}

// ---------------- pack W into WMMA B-operand layout ----------------
// B tile (32x16 f16): lane l holds col n = l&15, kbase = (l>>4)*8,
// element i (0..15) = W[kb*32 + kbase + (i&7) + (i>>3)*16][nt*16 + n].
// packed index: ((kb*16 + nt)*32 + lane)*16 + i  -> contiguous 32B per lane.
__global__ void packB_kernel(const float* __restrict__ W, _Float16* __restrict__ pB) {
    int idx  = blockIdx.x * 256 + threadIdx.x;       // 0 .. 32767
    int i    = idx & 15;
    int lane = (idx >> 4) & 31;
    int nt   = (idx >> 9) & 15;
    int kb   = idx >> 13;
    int k = kb * 32 + ((lane >> 4) * 8) + (i & 7) + ((i >> 3) * 16);
    int n = nt * 16 + (lane & 15);
    pB[idx] = (_Float16)W[k * OUT_F + n];
}

// ---------------- edge scatter: h[dst] += x[src] * w ----------------
// one wave per edge; lane handles a float4 chunk (32 lanes * 4 = 128 feats)
__global__ __launch_bounds__(256) void scatter_kernel(
    const float* __restrict__ x, const float* __restrict__ ew,
    const int* __restrict__ src, const int* __restrict__ dst,
    float* __restrict__ h) {
    int wave = blockIdx.x * 8 + (threadIdx.x >> 5);
    if (wave >= N_EDGES) return;
    int lane = threadIdx.x & 31;
    int s = src[wave];
    int d = dst[wave];
    float w = ew[wave];
    const float4* xr = (const float4*)(x + (size_t)s * IN_F);
    float4 v = xr[lane];
    float* hr = h + (size_t)d * IN_F + lane * 4;
    atomicAdd(hr + 0, v.x * w);
    atomicAdd(hr + 1, v.y * w);
    atomicAdd(hr + 2, v.z * w);
    atomicAdd(hr + 3, v.w * w);
}

// ------- WMMA GEMM [50000x128]x[128x256] + bias + ReLU + col stats -------
__global__ __launch_bounds__(256) void gemm_kernel(
    const float* __restrict__ h, const _Float16* __restrict__ pB,
    const float* __restrict__ bias, float* __restrict__ out,
    float* __restrict__ gsum, float* __restrict__ gsum2) {
    __shared__ float lsum[OUT_F];
    __shared__ float lsum2[OUT_F];
    int t = threadIdx.x;
    lsum[t] = 0.0f;
    lsum2[t] = 0.0f;
    __syncthreads();

    int wave = t >> 5;
    int lane = t & 31;
    int mt = blockIdx.x * 8 + wave;          // wave-uniform -> EXEC all-1s inside
    if (mt < M_TILES) {
        // ---- load A (16x128 f32 -> f16) in WMMA A layout ----
        // lane l: row = l&15; elems 0..7 = K kbase..kbase+7, 8..15 = K kbase+16..+23
        // where kbase = kb*32 + (l>>4)*8
        int r = mt * 16 + (lane & 15);
        int khalf = (lane >> 4) * 8;
        const float* arow = h + (size_t)r * IN_F;
        v16h a[4];
        #pragma unroll
        for (int kb = 0; kb < 4; ++kb) {
            int k0 = kb * 32 + khalf;
            #pragma unroll
            for (int i = 0; i < 8; ++i) {
                a[kb][i]     = (_Float16)arow[k0 + i];
                a[kb][i + 8] = (_Float16)arow[k0 + 16 + i];
            }
        }
        const v16h* pBv = (const v16h*)pB;
        int ncol = lane & 15;
        int rbase = mt * 16 + 8 * (lane >> 4);
        for (int nt = 0; nt < 16; ++nt) {
            v8f c = {};
            #pragma unroll
            for (int kb = 0; kb < 4; ++kb) {
                v16h bvec = pBv[(kb * 16 + nt) * 32 + lane];
                c = __builtin_amdgcn_wmma_f32_16x16x32_f16(
                        false, a[kb], false, bvec, (short)0, c, false, false);
            }
            // C/D layout: lane -> col (lane&15), VGPR j -> row j + 8*(lane>>4)
            int gcol = nt * 16 + ncol;
            float bv = bias[gcol];
            float s = 0.0f, s2 = 0.0f;
            #pragma unroll
            for (int j = 0; j < 8; ++j) {
                float v = c[j] + bv;
                v = v > 0.0f ? v : 0.0f;
                out[(size_t)(rbase + j) * OUT_F + gcol] = v;
                s += v;
                s2 += v * v;
            }
            atomicAdd(&lsum[gcol], s);
            atomicAdd(&lsum2[gcol], s2);
        }
    }
    __syncthreads();
    atomicAdd(&gsum[t], lsum[t]);
    atomicAdd(&gsum2[t], lsum2[t]);
}

// ---------------- batch stats -> scale/shift ----------------
__global__ void stats_kernel(const float* __restrict__ gsum,
                             const float* __restrict__ gsum2,
                             const float* __restrict__ gamma,
                             const float* __restrict__ beta,
                             float* __restrict__ scale,
                             float* __restrict__ shift) {
    int t = threadIdx.x;
    const float inv_n = 1.0f / (float)N_NODES;
    float mean = gsum[t] * inv_n;
    float var  = gsum2[t] * inv_n - mean * mean;
    float sc   = gamma[t] * rsqrtf(var + EPS);
    scale[t] = sc;
    shift[t] = beta[t] - mean * sc;
}

// ---------------- BN apply (float4 vectorized) ----------------
__global__ __launch_bounds__(256) void bn_kernel(float* __restrict__ out,
                                                 const float* __restrict__ scale,
                                                 const float* __restrict__ shift) {
    int idx = blockIdx.x * 256 + threadIdx.x;   // group of 4 floats
    float4 v = ((float4*)out)[idx];
    int col = (idx * 4) & (OUT_F - 1);
    float4 sc = *(const float4*)(scale + col);
    float4 sh = *(const float4*)(shift + col);
    v.x = v.x * sc.x + sh.x;
    v.y = v.y * sc.y + sh.y;
    v.z = v.z * sc.z + sh.z;
    v.w = v.w * sc.w + sh.w;
    ((float4*)out)[idx] = v;
}

extern "C" void kernel_launch(void* const* d_in, const int* in_sizes, int n_in,
                              void* d_out, int out_size, void* d_ws, size_t ws_size,
                              hipStream_t stream) {
    const float* x     = (const float*)d_in[0];   // [N, 128]
    const float* W     = (const float*)d_in[1];   // [128, 256]
    const float* b     = (const float*)d_in[2];   // [256]
    const float* gamma = (const float*)d_in[3];   // [256]
    const float* beta  = (const float*)d_in[4];   // [256]
    const float* ew    = (const float*)d_in[5];   // [E]
    const int*   src   = (const int*)d_in[6];     // [E]
    const int*   dst   = (const int*)d_in[7];     // [E]
    float*       out   = (float*)d_out;           // [N, 256]

    // workspace layout
    char* ws = (char*)d_ws;
    float*     h     = (float*)ws;                                  // N*128 f32 = 25.6 MB
    size_t off = (size_t)N_NODES * IN_F * sizeof(float);
    off = (off + 255) & ~(size_t)255;
    _Float16*  pB    = (_Float16*)(ws + off);                       // 128*256 f16 = 64 KB
    off += (size_t)IN_F * OUT_F * sizeof(_Float16);
    off = (off + 255) & ~(size_t)255;
    float*     gsum  = (float*)(ws + off);  off += OUT_F * sizeof(float);
    float*     gsum2 = (float*)(ws + off);  off += OUT_F * sizeof(float);
    float*     scale = (float*)(ws + off);  off += OUT_F * sizeof(float);
    float*     shift = (float*)(ws + off);  off += OUT_F * sizeof(float);

    // 1) zero h + accumulators (contiguous: h then gsum/gsum2 is not contiguous,
    //    so zero h and the 512 accumulators separately)
    {
        int nh = N_NODES * IN_F;
        zero_kernel<<<(nh + 255) / 256, 256, 0, stream>>>(h, nh);
        zero_kernel<<<2, 256, 0, stream>>>(gsum, 2 * OUT_F);  // gsum & gsum2 adjacent
    }
    // 2) pack W into WMMA B layout
    packB_kernel<<<(IN_F * OUT_F) / 256, 256, 0, stream>>>(W, pB);
    // 3) edge scatter
    scatter_kernel<<<(N_EDGES + 7) / 8, 256, 0, stream>>>(x, ew, src, dst, h);
    // 4) WMMA GEMM + bias + ReLU + stats
    gemm_kernel<<<(M_TILES + 7) / 8, 256, 0, stream>>>(h, pB, b, out, gsum, gsum2);
    // 5) stats -> scale/shift
    stats_kernel<<<1, OUT_F, 0, stream>>>(gsum, gsum2, gamma, beta, scale, shift);
    // 6) BN apply
    bn_kernel<<<(N_NODES * OUT_F / 4) / 256, 256, 0, stream>>>(out, scale, shift);
}
